// ZoeDepth_52536039964708
// MI455X (gfx1250) — matrix operand order repaired and checked by reading, not compiled, not run on hardware
//
#include <hip/hip_runtime.h>

typedef __attribute__((ext_vector_type(16))) _Float16 v16h;
typedef __attribute__((ext_vector_type(8)))  _Float16 v8h;
typedef __attribute__((ext_vector_type(8)))  float    v8f;

#define ALPHA_F 300.0f

// Problem dims (fixed by the reference)
constexpr int N_    = 4;
constexpr int C_    = 256;
constexpr int H_    = 128;
constexpr int W_    = 128;
constexpr int MLP   = 128;
constexpr int NATT  = 16;
constexpr int NBINS = 64;
constexpr int HW    = H_ * W_;                 // 16384
constexpr int TILES = N_ * H_ * (W_ / 16);     // 4096 (16-pixel tiles along W)
constexpr int WPB   = 8;                       // waves per block (256 threads)
constexpr int BLOCKS = TILES / WPB;            // 512

__global__ __launch_bounds__(256)
void zoedepth_attractor_fused(const float* __restrict__ x,
                              const float* __restrict__ bprev,
                              const float* __restrict__ w1,
                              const float* __restrict__ b1,
                              const float* __restrict__ w2,
                              const float* __restrict__ b2,
                              float* __restrict__ out0,
                              float* __restrict__ out1) {
    // LDS: 64KB + 4KB + 32KB + 8KB = 108KB / workgroup (CDNA5 WGP has 320KB)
    __shared__ _Float16 w1s[MLP * C_];          // [o][c] row-major, f16
    __shared__ _Float16 w2s[NATT * MLP];        // [a][m] row-major, f16
    __shared__ _Float16 hs[WPB][16 * MLP];      // per-wave hfeat scratch [pixel][m]
    __shared__ float    as_[WPB][NATT * 16];    // per-wave A scratch [attr][pixel]

    const int tid = threadIdx.x;

    const int wave = tid >> 5;
    const int lane = tid & 31;
    const int half = lane >> 4;     // which half-wave (selects K sub-block in frags)
    const int lr   = lane & 15;     // pixel index (A/C rows) and N column (B/C cols)

    const int tile = blockIdx.x * WPB + wave;   // exactly TILES waves launched
    const int n    = tile >> 10;                // tiles per image = 128*8 = 1024
    const int rem  = tile & 1023;
    const int h    = rem >> 3;
    const int w0   = (rem & 7) << 4;

    // Early prefetch of this tile's b_prev rows (64 rows x 64B): overlap the
    // HBM latency of the attractor stream with the WMMA phase. Each lane
    // targets one bin row -> 2 instructions cover all 64 rows (gfx1250
    // global_prefetch_b8).
    {
        const size_t pbase = (size_t)n * NBINS * HW + (size_t)h * W_ + w0;
        __builtin_prefetch(&bprev[pbase + (size_t)lane        * HW], 0, 3);
        __builtin_prefetch(&bprev[pbase + (size_t)(lane + 32) * HW], 0, 3);
    }

    // ---- Phase 0: stage weights into LDS, converting f32 -> f16 ----
    #pragma unroll 4
    for (int i = tid; i < MLP * C_; i += 256) w1s[i] = (_Float16)w1[i];
    for (int i = tid; i < NATT * MLP; i += 256) w2s[i] = (_Float16)w2[i];
    __syncthreads();

    // ---- GEMM1: hfeat[16px][128] = relu(x_tile (16x256) * w1^T + b1) ----
    v8f acc[8];
    {
        v8f z = {};
        #pragma unroll
        for (int j = 0; j < 8; ++j) acc[j] = z;
    }

    const size_t xbase = (size_t)n * C_ * HW + (size_t)h * W_ + w0;

    #pragma unroll
    for (int k = 0; k < 8; ++k) {               // K chunks of 32 channels
        // Build A fragment (16x32, f16). ISA layout: lane<16 holds K base 0,
        // lane>=16 holds K base 8; elements 0..7 -> K+e, 8..15 -> K+e+8.
        v16h a;
        const int cbase = k * 32 + half * 8;
        #pragma unroll
        for (int e = 0; e < 16; ++e) {
            const int c = cbase + ((e < 8) ? e : (e + 8));
            a[e] = (_Float16)x[xbase + (size_t)c * HW + lr];
        }
        // Batch-preload all 8 B fragments for this K chunk (16x ds_load_b128)
        // so the WMMA pipe doesn't lock-step on per-fragment dscnt waits.
        union { v16h v; v8h h2[2]; } bu[8];
        #pragma unroll
        for (int j = 0; j < 8; ++j) {
            const int o = j * 16 + lr;          // B: N = lane%16
            // B layout: same K order as A: elems 0..7 -> K=8*half+e, 8..15 -> K=16+8*half+e
            const v8h* bp = (const v8h*)&w1s[o * C_ + k * 32 + half * 8];
            bu[j].h2[0] = bp[0];
            bu[j].h2[1] = bp[2];
        }
        #pragma unroll
        for (int j = 0; j < 8; ++j) {           // 8 output-column tiles of 16
            acc[j] = __builtin_amdgcn_wmma_f32_16x16x32_f16(
                false, a, false, bu[j].v, (short)0, acc[j], false, false);
        }
    }

    // bias + ReLU; spill to per-wave LDS as f16 in [pixel][m] for re-layout
    #pragma unroll
    for (int j = 0; j < 8; ++j) {
        const int   o    = j * 16 + lr;
        const float bias = b1[o];
        #pragma unroll
        for (int r = 0; r < 8; ++r) {
            const int px = r + half * 8;        // C layout: M = r + 8*(lane/16)
            float v = acc[j][r] + bias;
            v = fmaxf(v, 0.0f);
            hs[wave][px * MLP + o] = (_Float16)v;
        }
    }
    // per-wave LDS ops are hardware in-order; keep compiler from reordering
    asm volatile("" ::: "memory");

    // ---- GEMM2: A_pre[16px][16att] = hfeat (16x128) * w2^T ----
    v8f acc2 = {};
    {
        union { v16h v; v8h h2[2]; } au[4], bu[4];
        #pragma unroll
        for (int k = 0; k < 4; ++k) {
            const int mbase = k * 32 + half * 8;
            const v8h* ap = (const v8h*)&hs[wave][lr * MLP + mbase];
            au[k].h2[0] = ap[0];                 // m = mbase + 0..7
            au[k].h2[1] = ap[2];                 // m = mbase + 16..23
            const v8h* bp2 = (const v8h*)&w2s[lr * MLP + k * 32 + half * 8];
            bu[k].h2[0] = bp2[0];
            bu[k].h2[1] = bp2[2];
        }
        #pragma unroll
        for (int k = 0; k < 4; ++k) {
            acc2 = __builtin_amdgcn_wmma_f32_16x16x32_f16(
                false, au[k].v, false, bu[k].v, (short)0, acc2, false, false);
        }
    }

    // bias + softplus; stage A in LDS as [attr][pixel] (conflict-free reads)
    {
        const float bias2 = b2[lr];              // attr = lane%16 (C-layout N)
        #pragma unroll
        for (int r = 0; r < 8; ++r) {
            const int px = r + half * 8;
            const float v  = acc2[r] + bias2;
            const float sp = fmaxf(v, 0.0f) + __logf(1.0f + __expf(-fabsf(v)));
            as_[wave][lr * 16 + px] = sp;
        }
    }
    asm volatile("" ::: "memory");

    // ---- Attractor stage: per (pixel=lr, bin) ----
    float Aval[NATT];
    #pragma unroll
    for (int a = 0; a < NATT; ++a) Aval[a] = as_[wave][a * 16 + lr];

    const size_t bbase = (size_t)n * NBINS * HW + (size_t)h * W_ + w0 + lr;
    #pragma unroll 4
    for (int it = 0; it < 32; ++it) {
        const int    bin = half * 32 + it;       // half-wave walks its 32 bins
        const size_t off = bbase + (size_t)bin * HW;  // coalesced 64B per half-wave
        const float  bc  = bprev[off];
        float s = bc;
        #pragma unroll
        for (int a = 0; a < NATT; ++a) {
            const float dx = Aval[a] - bc;
            s += __expf(-ALPHA_F * dx * dx) * dx;   // v_exp_f32 (TRANS)
        }
        out0[off] = s;
        out1[off] = s;   // reference returns the tuple (b_new, b_new)
    }
}

extern "C" void kernel_launch(void* const* d_in, const int* in_sizes, int n_in,
                              void* d_out, int out_size, void* d_ws, size_t ws_size,
                              hipStream_t stream) {
    const float* x  = (const float*)d_in[0];
    const float* bp = (const float*)d_in[1];
    const float* w1 = (const float*)d_in[2];
    const float* b1 = (const float*)d_in[3];
    const float* w2 = (const float*)d_in[4];
    const float* b2 = (const float*)d_in[5];
    float* out = (float*)d_out;
    const size_t halfOut = (size_t)N_ * NBINS * H_ * W_;   // 4,194,304 elements
    zoedepth_attractor_fused<<<BLOCKS, 256, 0, stream>>>(
        x, bp, w1, b1, w2, b2, out, out + halfOut);
    (void)in_sizes; (void)n_in; (void)out_size; (void)d_ws; (void)ws_size;
}
